// LSTMLayer_31207232373232
// MI455X (gfx1250) — compile-verified, matrix operand order checked by reading, not run on hardware
//
#include <hip/hip_runtime.h>
#include <cstdint>
#include <cstddef>

// ---------------------------------------------------------------------------
// LSTM layer for MI455X (gfx1250): bf16 WMMA matmuls, fp32 accumulate/state.
//   B=128, T=1024, D=768, H=768, 4H=3072
// - A tiles staged into LDS with GLOBAL_LOAD_ASYNC_TO_LDS_B128 (ASYNCcnt),
//   shared by the 8 waves of a block; 776-elem pitch -> conflict-free reads.
// - W global stream triple-buffered (distance-2 prefetch), A double-buffered;
//   __builtin_amdgcn_sched_barrier pins the software pipeline so v_wmma
//   overlaps 16+ in-flight global_load_b128.
// - Recurrence runs in ONE persistent kernel (48 co-resident blocks) with a
//   grid-wide arrival counter between timesteps: no 1024-launch overhead.
// ---------------------------------------------------------------------------

#define BB 128
#define TT 1024
#define DD 768
#define HH 768
#define G4 3072          // 4*H
#define NK 24            // 768 / 32 K-chunks
#define APAD 776         // 768 + 8 bf16 row padding (388 dwords % 64 == 4)
#define NBLK 48          // persistent grid size (8 b-tiles x 6 h-groups)

typedef __attribute__((ext_vector_type(16))) __bf16       v16bf;
typedef __attribute__((ext_vector_type(8)))  float        v8f;
typedef __attribute__((ext_vector_type(4)))  unsigned int u32x4;

union FragBF {
    v16bf v;
    u32x4 q[2];
};

__device__ __forceinline__ unsigned short f2bf(float f) {
    unsigned u = __float_as_uint(f);
    u += 0x7FFFu + ((u >> 16) & 1u);   // round-to-nearest-even
    return (unsigned short)(u >> 16);
}

__device__ __forceinline__ float fast_sigmoid(float x) {
    return 1.0f / (1.0f + __expf(-x));
}

// Load 4 B-fragments (one 16-col tile each) at K-offset `off` (elements).
__device__ __forceinline__ void load_w4(FragBF* w,
                                        const unsigned short* r0,
                                        const unsigned short* r1,
                                        const unsigned short* r2,
                                        const unsigned short* r3, int off) {
    w[0].q[0] = *(const u32x4*)(r0 + off);
    w[0].q[1] = *(const u32x4*)(r0 + off + 16);
    w[1].q[0] = *(const u32x4*)(r1 + off);
    w[1].q[1] = *(const u32x4*)(r1 + off + 16);
    w[2].q[0] = *(const u32x4*)(r2 + off);
    w[2].q[1] = *(const u32x4*)(r2 + off + 16);
    w[3].q[0] = *(const u32x4*)(r3 + off);
    w[3].q[1] = *(const u32x4*)(r3 + off + 16);
}

__device__ __forceinline__ void load_a(FragBF* a, const unsigned short* ap, int k) {
    a->q[0] = *(const u32x4*)(ap + k);
    a->q[1] = *(const u32x4*)(ap + k + 16);
}

// ---------------------------------------------------------------------------
// fp32 -> bf16 bulk conversion (grid-stride)
// ---------------------------------------------------------------------------
__global__ void k_f32_to_bf16(const float* __restrict__ src,
                              unsigned short* __restrict__ dst, size_t n) {
    size_t i  = (size_t)blockIdx.x * blockDim.x + threadIdx.x;
    size_t st = (size_t)gridDim.x * blockDim.x;
    for (; i < n; i += st) dst[i] = f2bf(src[i]);
}

// ---------------------------------------------------------------------------
// setup: h0 -> bf16 buffer, c0 -> fp32 state, bias = bias_ih + bias_hh,
//        zero the persistent-kernel sync counter (deterministic per call)
// ---------------------------------------------------------------------------
__global__ void k_setup(const float* __restrict__ h0, const float* __restrict__ c0,
                        const float* __restrict__ bih, const float* __restrict__ bhh,
                        unsigned short* __restrict__ hbuf, float* __restrict__ cbuf,
                        float* __restrict__ bias, unsigned* __restrict__ sync) {
    int i = blockIdx.x * blockDim.x + threadIdx.x;
    if (i < BB * HH) {
        hbuf[i] = f2bf(h0[i]);
        cbuf[i] = c0[i];
    }
    if (i < G4) bias[i] = bih[i] + bhh[i];
    if (i == 0) *sync = 0u;
}

// ---------------------------------------------------------------------------
// Cooperative A-tile staging via CDNA5 async memory->LDS DMA (ASYNCcnt).
// 16 rows x 768 bf16 into LDS (pitch APAD); 256 threads x 6 coalesced 16B
// chunks. LDS dest address = low 32 bits of the generic pointer (AS3 offset).
// ---------------------------------------------------------------------------
__device__ __forceinline__ void stage_A_async(unsigned short* sA,
                                              const unsigned short* __restrict__ g,
                                              size_t row_stride, size_t base) {
    const int tid = threadIdx.x;
#pragma unroll
    for (int i = 0; i < 6; ++i) {
        const int chunk = tid + i * 256;       // 0..1535
        const int row   = chunk / 96;
        const int c16   = chunk % 96;
        const unsigned short* src = g + base + (size_t)row * row_stride + c16 * 8;
        const unsigned dst = (unsigned)(uintptr_t)(sA + row * APAD + c16 * 8);
        asm volatile("global_load_async_to_lds_b128 %0, %1, off"
                     :: "v"(dst), "v"(src) : "memory");
    }
    asm volatile("s_wait_asynccnt 0x0" ::: "memory");
    __syncthreads();
}

// ---------------------------------------------------------------------------
// Phase 1: x_proj[t,b,g] = sum_d x[b,t,d]*W_ih[g,d] + bias[g]
// Block = one 16-row tile (t, b0) x 512 gate-cols (8 waves x 64 cols).
// ---------------------------------------------------------------------------
__global__ void __launch_bounds__(256)
k_xproj(const unsigned short* __restrict__ xbf,   // [B,T,D] bf16
        const unsigned short* __restrict__ wih,   // [4H,D]  bf16
        const float* __restrict__ bias,           // [4H]
        float* __restrict__ xproj) {              // [T,B,4H] fp32
    __shared__ unsigned short sA[16 * APAD];

    const int bid = blockIdx.x;
    const int mt  = bid / 6;                 // 0..8191 row tiles
    const int ng  = bid % 6;                 // col group (512 cols)
    const int t   = mt >> 3;
    const int b0  = (mt & 7) << 4;
    const int wid  = threadIdx.x >> 5;
    const int lane = threadIdx.x & 31;
    const int n0   = (ng * 8 + wid) << 6;    // 64-col strip per wave
    const int lrow = lane & 15;
    const int hi   = lane >> 4;

    stage_A_async(sA, xbf, (size_t)TT * DD, ((size_t)b0 * TT + t) * (size_t)DD);

    const unsigned short* br0 = wih + (size_t)(n0 +  0 + lrow) * DD;
    const unsigned short* br1 = wih + (size_t)(n0 + 16 + lrow) * DD;
    const unsigned short* br2 = wih + (size_t)(n0 + 32 + lrow) * DD;
    const unsigned short* br3 = wih + (size_t)(n0 + 48 + lrow) * DD;
    const unsigned short* ap  = sA + lrow * APAD + hi * 8;

    v8f acc0 = {}, acc1 = {}, acc2 = {}, acc3 = {};
    FragBF w0[4], w1[4], w2[4], aA, aB;
    load_w4(w0, br0, br1, br2, br3, hi * 8);          // prologue: chunks 0,1
    load_w4(w1, br0, br1, br2, br3, 32 + hi * 8);
    load_a(&aA, ap, 0);
    __builtin_amdgcn_sched_barrier(0);

#pragma unroll
    for (int kk = 0; kk < NK; ++kk) {
        FragBF* wc = ((kk % 3) == 0) ? w0 : ((kk % 3) == 1) ? w1 : w2;
        FragBF* wp = (((kk + 2) % 3) == 0) ? w0 : (((kk + 2) % 3) == 1) ? w1 : w2;
        FragBF* ac = (kk & 1) ? &aB : &aA;
        FragBF* an = (kk & 1) ? &aA : &aB;
        if (kk + 2 < NK)                               // distance-2 W prefetch
            load_w4(wp, br0, br1, br2, br3, (kk + 2) * 32 + hi * 8);
        if (kk + 1 < NK)                               // distance-1 A prefetch
            load_a(an, ap, (kk + 1) * 32);
        __builtin_amdgcn_sched_barrier(0);             // keep prefetch ABOVE WMMAs
        acc0 = __builtin_amdgcn_wmma_f32_16x16x32_bf16(false, ac->v, false, wc[0].v,
                                                       (short)0, acc0, false, false);
        acc1 = __builtin_amdgcn_wmma_f32_16x16x32_bf16(false, ac->v, false, wc[1].v,
                                                       (short)0, acc1, false, false);
        acc2 = __builtin_amdgcn_wmma_f32_16x16x32_bf16(false, ac->v, false, wc[2].v,
                                                       (short)0, acc2, false, false);
        acc3 = __builtin_amdgcn_wmma_f32_16x16x32_bf16(false, ac->v, false, wc[3].v,
                                                       (short)0, acc3, false, false);
        __builtin_amdgcn_sched_barrier(0);
    }

    const int col0 = n0 + lrow;
    const float bi0 = bias[col0 +  0];
    const float bi1 = bias[col0 + 16];
    const float bi2 = bias[col0 + 32];
    const float bi3 = bias[col0 + 48];
#pragma unroll
    for (int r = 0; r < 8; ++r) {
        const int b = b0 + r + hi * 8;
        float* row = xproj + ((size_t)t * BB + b) * (size_t)G4;
        row[col0 +  0] = acc0[r] + bi0;
        row[col0 + 16] = acc1[r] + bi1;
        row[col0 + 32] = acc2[r] + bi2;
        row[col0 + 48] = acc3[r] + bi3;
    }
}

// ---------------------------------------------------------------------------
// Grid-wide split barrier for the persistent recurrence. Monotonic counter:
// step t completes when counter reaches (t+1)*NBLK. Release fence before
// arrival; block-wide acquire fence after, so re-staged h reads (and the
// async-to-LDS loads) cannot hit stale cache lines. No reset -> no races.
// ---------------------------------------------------------------------------
__device__ __forceinline__ void grid_sync(unsigned* cnt, unsigned target) {
    __syncthreads();
    if (threadIdx.x == 0) {
        __threadfence();                       // release h stores
        atomicAdd(cnt, 1u);
        while (*(volatile unsigned*)cnt < target)
            __builtin_amdgcn_s_sleep(2);
    }
    __syncthreads();
    __threadfence();                           // block-wide acquire
}

// ---------------------------------------------------------------------------
// Phase 2 (persistent): all 1024 timesteps in one launch.
// Block = one 16-batch tile x 8 h-tiles (one per wave); the 8 waves share the
// async-staged h tile in LDS. Each wave accumulates ALL FOUR gates (cols
// g*768+n0) and does the LSTM nonlinearity + c/h update in-register. x_proj
// tail reads are issued BEFORE the K loop so HBM latency hides behind the
// WMMA pipeline. W_hh (4.7 MB bf16) stays L2-resident across all steps.
// ---------------------------------------------------------------------------
__global__ void __launch_bounds__(256)
k_lstm_seq(const unsigned short* __restrict__ whh,   // [4H,H] bf16
           const float* __restrict__ xproj,          // [T,B,4H] fp32 (incl. biases)
           unsigned short* __restrict__ hb0,         // [B,H] bf16 (t even reads)
           unsigned short* __restrict__ hb1,         // [B,H] bf16 (t odd  reads)
           float* __restrict__ cbuf,                 // [B,H] fp32 (in-place)
           float* __restrict__ out,                  // [B,T,H] fp32
           float* __restrict__ hT, float* __restrict__ cT,
           unsigned* __restrict__ sync) {
    __shared__ unsigned short sA[16 * APAD];

    const int bid = blockIdx.x;                  // 0..47
    const int mt  = bid & 7;                     // b tile
    const int ng  = bid >> 3;                    // h-tile group (0..5)
    const int b0  = mt << 4;
    const int wid  = threadIdx.x >> 5;
    const int lane = threadIdx.x & 31;
    const int n0   = (ng * 8 + wid) << 4;        // 16-col h tile per wave
    const int lrow = lane & 15;
    const int hi   = lane >> 4;
    const int h    = n0 + lrow;

    const unsigned short* wi = whh + (size_t)(0 * HH + n0 + lrow) * HH;
    const unsigned short* wf = whh + (size_t)(1 * HH + n0 + lrow) * HH;
    const unsigned short* wg = whh + (size_t)(2 * HH + n0 + lrow) * HH;
    const unsigned short* wo = whh + (size_t)(3 * HH + n0 + lrow) * HH;
    const unsigned short* ap = sA + lrow * APAD + hi * 8;

    for (int t = 0; t < TT; ++t) {
        const unsigned short* hin = (t & 1) ? hb1 : hb0;
        unsigned short*       ho  = (t & 1) ? hb0 : hb1;

        __syncthreads();                          // LDS reuse safe across steps
        stage_A_async(sA, hin, (size_t)HH, (size_t)b0 * HH);

        // Early-issue x_proj tail reads (consumed after the K loop).
        float xpv[4][8];
#pragma unroll
        for (int r = 0; r < 8; ++r) {
            const int b = b0 + r + hi * 8;
            const float* xp = xproj + ((size_t)t * BB + b) * (size_t)G4;
            xpv[0][r] = xp[h];
            xpv[1][r] = xp[HH + h];
            xpv[2][r] = xp[2 * HH + h];
            xpv[3][r] = xp[3 * HH + h];
        }
        __builtin_amdgcn_sched_barrier(0);

        v8f ai = {}, af = {}, ag = {}, ao = {};
        FragBF w0[4], w1[4], w2[4], aA, aB;
        load_w4(w0, wi, wf, wg, wo, hi * 8);     // prologue: chunks 0,1
        load_w4(w1, wi, wf, wg, wo, 32 + hi * 8);
        load_a(&aA, ap, 0);
        __builtin_amdgcn_sched_barrier(0);

#pragma unroll
        for (int kk = 0; kk < NK; ++kk) {
            FragBF* wc = ((kk % 3) == 0) ? w0 : ((kk % 3) == 1) ? w1 : w2;
            FragBF* wp = (((kk + 2) % 3) == 0) ? w0 : (((kk + 2) % 3) == 1) ? w1 : w2;
            FragBF* ac = (kk & 1) ? &aB : &aA;
            FragBF* an = (kk & 1) ? &aA : &aB;
            if (kk + 2 < NK)                     // distance-2 W prefetch
                load_w4(wp, wi, wf, wg, wo, (kk + 2) * 32 + hi * 8);
            if (kk + 1 < NK)                     // distance-1 A prefetch
                load_a(an, ap, (kk + 1) * 32);
            __builtin_amdgcn_sched_barrier(0);   // keep prefetch ABOVE WMMAs
            ai = __builtin_amdgcn_wmma_f32_16x16x32_bf16(false, ac->v, false, wc[0].v,
                                                         (short)0, ai, false, false);
            af = __builtin_amdgcn_wmma_f32_16x16x32_bf16(false, ac->v, false, wc[1].v,
                                                         (short)0, af, false, false);
            ag = __builtin_amdgcn_wmma_f32_16x16x32_bf16(false, ac->v, false, wc[2].v,
                                                         (short)0, ag, false, false);
            ao = __builtin_amdgcn_wmma_f32_16x16x32_bf16(false, ac->v, false, wc[3].v,
                                                         (short)0, ao, false, false);
            __builtin_amdgcn_sched_barrier(0);
        }

#pragma unroll
        for (int r = 0; r < 8; ++r) {
            const int b = b0 + r + hi * 8;
            const float iv = fast_sigmoid(ai[r] + xpv[0][r]);
            const float fv = fast_sigmoid(af[r] + xpv[1][r]);
            const float gv = tanhf(ag[r] + xpv[2][r]);
            const float ov = fast_sigmoid(ao[r] + xpv[3][r]);

            const size_t idx = (size_t)b * HH + h;
            const float cy = fv * cbuf[idx] + iv * gv;
            const float hy = ov * tanhf(cy);
            cbuf[idx] = cy;
            ho[idx]   = f2bf(hy);
            out[((size_t)b * TT + t) * HH + h] = hy;
            if (t == TT - 1) {   // emit tuple outputs (hT, cT)
                hT[idx] = hy;
                cT[idx] = cy;
            }
        }

        grid_sync(sync, (unsigned)(t + 1) * NBLK);
    }
}

// ---------------------------------------------------------------------------
// Host-side orchestration (all launches on `stream`; graph-capture safe)
// ---------------------------------------------------------------------------
extern "C" void kernel_launch(void* const* d_in, const int* in_sizes, int n_in,
                              void* d_out, int out_size, void* d_ws, size_t ws_size,
                              hipStream_t stream) {
    const float* x   = (const float*)d_in[0];   // [B,T,D]
    const float* h0  = (const float*)d_in[1];   // [1,B,H]
    const float* c0  = (const float*)d_in[2];   // [1,B,H]
    const float* wih = (const float*)d_in[3];   // [4H,D]
    const float* whh = (const float*)d_in[4];   // [4H,H]
    const float* bih = (const float*)d_in[5];   // [4H]
    const float* bhh = (const float*)d_in[6];   // [4H]

    // Workspace carve (256B aligned):
    char* p = (char*)d_ws;
    auto carve = [&p](size_t bytes) -> char* {
        char* r = p;
        p += (bytes + 255) & ~(size_t)255;
        return r;
    };
    unsigned short* xbf  = (unsigned short*)carve((size_t)BB * TT * DD * 2); // 201 MB
    unsigned short* wihb = (unsigned short*)carve((size_t)G4 * DD * 2);      // 4.7 MB
    unsigned short* whhb = (unsigned short*)carve((size_t)G4 * HH * 2);      // 4.7 MB
    unsigned short* hb0  = (unsigned short*)carve((size_t)BB * HH * 2);
    unsigned short* hb1  = (unsigned short*)carve((size_t)BB * HH * 2);
    float* cbuf    = (float*)carve((size_t)BB * HH * 4);
    float* bias    = (float*)carve((size_t)G4 * 4);
    unsigned* sync = (unsigned*)carve(256);
    float* xproj   = (float*)carve((size_t)TT * BB * G4 * 4);                // 1.6 GB

    // Phase 0: conversions + state init (+ zero sync counter every call)
    k_f32_to_bf16<<<4096, 256, 0, stream>>>(x, xbf, (size_t)BB * TT * DD);
    k_f32_to_bf16<<<1024, 256, 0, stream>>>(wih, wihb, (size_t)G4 * DD);
    k_f32_to_bf16<<<1024, 256, 0, stream>>>(whh, whhb, (size_t)G4 * HH);
    k_setup<<<(BB * HH + 255) / 256, 256, 0, stream>>>(h0, c0, bih, bhh,
                                                       hb0, cbuf, bias, sync);

    // Phase 1: input projection GEMM (8192 row-tiles x 6 col-groups)
    k_xproj<<<49152, 256, 0, stream>>>(xbf, wihb, bias, xproj);

    // Phase 2: single persistent launch, 1024 internally-synchronized steps
    float* out = (float*)d_out;
    float* hT  = out + (size_t)BB * TT * HH;
    float* cT  = hT + (size_t)BB * HH;
    k_lstm_seq<<<NBLK, 256, 0, stream>>>(whhb, xproj, hb0, hb1, cbuf,
                                         out, hT, cT, sync);
}
